// Decoder_interpolation_63788854280413
// MI455X (gfx1250) — matrix operand order repaired and checked by reading, not compiled, not run on hardware
//
#include <hip/hip_runtime.h>

// ---------------------------------------------------------------------------
// MI455X (gfx1250) fused decoder: bilinear feature gather + residual MLP,
// GEMM chain on v_wmma_f32_16x16x32_bf16 (wave32, M=16 per wave).
// ---------------------------------------------------------------------------

typedef __attribute__((ext_vector_type(16))) __bf16 v16bf;
typedef __attribute__((ext_vector_type(8)))  float  v8f;

#define T_PTS   30000
#define L_VIEWS 6
#define GH      128
#define GW      128
#define CD      32
#define WAVES   4

union U16bf { unsigned int u[8]; v16bf v; };

// Native RNE f32 -> bf16 (lets the backend use v_cvt bf16 converts).
__device__ __forceinline__ unsigned short bfb(float f) {
    __bf16 h = (__bf16)f;
    return __builtin_bit_cast(unsigned short, h);
}
__device__ __forceinline__ unsigned int packbf(float lo, float hi) {
    return (unsigned int)bfb(lo) | ((unsigned int)bfb(hi) << 16);
}
__device__ __forceinline__ unsigned int relu_pk(unsigned int u) {
    unsigned int lo = (u & 0x8000u)     ? 0u : (u & 0x0000FFFFu);
    unsigned int hi = (u & 0x80000000u) ? 0u : (u & 0xFFFF0000u);
    return lo | hi;
}
__device__ __forceinline__ v8f wmma_bf16(v16bf a, v16bf b, v8f c) {
    return __builtin_amdgcn_wmma_f32_16x16x32_bf16(false, a, false, b, (short)0, c, false, false);
}

// B fragment (K x 16 tile, bf16) preconverted in global ws: 8 uints/lane/tile.
__device__ __forceinline__ v16bf loadBFragG(const unsigned int* __restrict__ F, int tile, int lane) {
    const uint4* q = (const uint4*)(F + (tile * 32 + lane) * 8);
    uint4 a = q[0], b = q[1];
    U16bf u;
    u.u[0] = a.x; u.u[1] = a.y; u.u[2] = a.z; u.u[3] = a.w;
    u.u[4] = b.x; u.u[5] = b.y; u.u[6] = b.z; u.u[7] = b.w;
    return u.v;
}

// A fragment raw uints from LDS bf16 row-major activation tile.
// A 16x32 layout (matches B frags): VGPR j -> K = hsel*16 + 2*j + {0,1}
// contiguous 8 VGPRs -> two ds_load_b128 per lane.
__device__ __forceinline__ void loadRaw8(const unsigned short* baseHalf, int row, int rowU,
                                         int kt, int hsel, unsigned int r[8]) {
    const unsigned int* bu = (const unsigned int*)baseHalf + row * rowU + kt * 16 + hsel * 8;
    const uint4* q = (const uint4*)bu;
    uint4 a = q[0], b = q[1];
    r[0] = a.x; r[1] = a.y; r[2] = a.z; r[3] = a.w;
    r[4] = b.x; r[5] = b.y; r[6] = b.z; r[7] = b.w;
}
__device__ __forceinline__ v16bf mkA(const unsigned int r[8]) {
    U16bf u;
#pragma unroll
    for (int j = 0; j < 8; ++j) u.u[j] = r[j];
    return u.v;
}
__device__ __forceinline__ v16bf mkA_relu(const unsigned int r[8]) {
    U16bf u;
#pragma unroll
    for (int j = 0; j < 8; ++j) u.u[j] = relu_pk(r[j]);
    return u.v;
}

// Store C-layout pair (16x32) as bf16 into LDS staging [16 rows][32 ch].
__device__ __forceinline__ void storeC_bf16(unsigned short* xw, v8f c0, v8f c1, int mIdx, int hsel) {
#pragma unroll
    for (int v = 0; v < 8; ++v) {
        int row = v + 8 * hsel;
        xw[row * 32 + mIdx]      = bfb(c0[v]);
        xw[row * 32 + 16 + mIdx] = bfb(c1[v]);
    }
}
// Same, but stages relu(x) (consumer then loads raw).
__device__ __forceinline__ void storeC_relu_bf16(unsigned short* xw, v8f c0, v8f c1, int mIdx, int hsel) {
#pragma unroll
    for (int v = 0; v < 8; ++v) {
        int row = v + 8 * hsel;
        xw[row * 32 + mIdx]      = bfb(fmaxf(c0[v], 0.0f));
        xw[row * 32 + 16 + mIdx] = bfb(fmaxf(c1[v], 0.0f));
    }
}

#define DS_WAIT() asm volatile("s_wait_dscnt 0" ::: "memory")

// ---------------------------------------------------------------------------
// Prep: convert [N][K] fp32 weights into bf16 B-fragment order in global ws.
// tile = (mat*ktiles + kt)*ntiles + nt ; per tile: 32 lanes x 8 uints.
// B layout: lane gives N = nt*16 + (lane&15); K = kt*32 + (lane>=16?16:0) + half.
// ---------------------------------------------------------------------------
__device__ void buildB(unsigned int* dst, const float* W, int Kdim,
                       int ntiles, int ktiles, int nmats) {
    const int total = nmats * ktiles * ntiles * 256;
    const int Nmat = ntiles * 16;
    for (int idx = threadIdx.x; idx < total; idx += blockDim.x) {
        int j = idx & 7, lane = (idx >> 3) & 31, tile = idx >> 8;
        int nt = tile % ntiles;
        int r = tile / ntiles;
        int kt = r % ktiles;
        int mat = r / ktiles;
        int n = nt * 16 + (lane & 15);
        int k0 = kt * 32 + ((lane >> 4) * 16) + 2 * j;
        const float* Wm = W + (size_t)mat * Nmat * Kdim + (size_t)n * Kdim;
        dst[idx] = packbf(Wm[k0], Wm[k0 + 1]);
    }
}

__global__ void prep_kernel(const float* __restrict__ w0, const float* __restrict__ ws,
                            const float* __restrict__ w1b0, const float* __restrict__ bw0,
                            const float* __restrict__ bw1, unsigned int* __restrict__ dst) {
    buildB(dst,         w0,   256, 2, 8, 1);   // blk0_w0 : 16 tiles @ 0
    buildB(dst + 4096,  ws,   256, 2, 8, 1);   // blk0_ws : 16 tiles @ 4096
    buildB(dst + 8192,  w1b0, 32,  2, 1, 1);   // blk0_w1 :  2 tiles @ 8192
    buildB(dst + 8704,  bw0,  32,  2, 1, 4);   // blk_w0  :  8 tiles @ 8704
    buildB(dst + 10752, bw1,  32,  2, 1, 4);   // blk_w1  :  8 tiles @ 10752
}

// ---------------------------------------------------------------------------
// Main fused kernel: one wave per 16-point tile.
// ---------------------------------------------------------------------------
__global__ __launch_bounds__(WAVES * 32) void decoder_kernel(
    const float* __restrict__ p, const float* __restrict__ c, const float* __restrict__ Cmat,
    const float* __restrict__ fc_p_w, const float* __restrict__ fc_p_b,
    const float* __restrict__ blk0_b0, const float* __restrict__ blk0_b1,
    const float* __restrict__ blk_b0, const float* __restrict__ blk_b1,
    const float* __restrict__ fc_out_w, const float* __restrict__ fc_out_b,
    const unsigned int* __restrict__ wfrag, float* __restrict__ out, int nPts) {

    __shared__ __align__(16) unsigned short netS[WAVES * 4096]; // bf16 [16][256] per wave
    __shared__ __align__(16) unsigned short xS[WAVES * 512];    // bf16 [16][32]  per wave
    __shared__ float featS[WAVES * 512];                        // f32  [16][32]  per wave

    const int lane = threadIdx.x & 31;
    const int wv   = threadIdx.x >> 5;
    const int mIdx = lane & 15;     // row (point) for A / N-channel for B,C
    const int hsel = lane >> 4;     // lane-half select
    const int ptBase = (blockIdx.x * WAVES + wv) * 16;
    const int b = ptBase / T_PTS;   // 16-tiles never straddle batches (30000 % 16 == 0)

    unsigned short* nS = netS + wv * 4096;
    unsigned short* xw = xS + wv * 512;
    float* fS = featS + wv * 512;

    // ---- point load (each lane: point mIdx of the tile) ----
    int ptIdx = ptBase + mIdx; if (ptIdx >= nPts) ptIdx = nPts - 1;
    const float* pm = p + (size_t)ptIdx * 3;
    float px = pm[0], py = pm[1], pz = pm[2];

    // ---- bilinear multi-view gather: lane handles 16 channels (hsel half) ----
    const float inv_md = 1.0f / 0.55f;
    float qx = px * inv_md, qy = py * inv_md, qz = pz * inv_md;
    float acc[16];
#pragma unroll
    for (int k = 0; k < 16; ++k) acc[k] = 0.0f;
    const float* cb = c + (size_t)b * L_VIEWS * GH * GW * CD;
#pragma unroll
    for (int vw = 0; vw < L_VIEWS; ++vw) {
        const float* Cm = Cmat + ((size_t)b * L_VIEWS + vw) * 12;
        float rden = 1.0f / (Cm[9] + 0.05f);
        float prx = (Cm[0] * qx + Cm[1] * qy + Cm[2] * qz) * rden;
        float pry = (Cm[3] * qx + Cm[4] * qy + Cm[5] * qz) * rden;
        float X = (prx + 1.0f) * 63.5f;     // / (2/(H-1))
        float Y = (pry + 1.0f) * 63.5f;
        float x0f = floorf(X), x1f = ceilf(X), y0f = floorf(Y), y1f = ceilf(Y);
        float dxw = x1f - X, dyw = y1f - Y;
        int x0 = min(max((int)x0f, 0), GH - 1), x1 = min(max((int)x1f, 0), GH - 1);
        int y0 = min(max((int)y0f, 0), GW - 1), y1 = min(max((int)y1f, 0), GW - 1);
        const float* base = cb + (size_t)vw * GH * GW * CD + hsel * 16;
        const float4* f11 = (const float4*)(base + ((size_t)x0 * GW + y0) * CD);
        const float4* f12 = (const float4*)(base + ((size_t)x1 * GW + y0) * CD);
        const float4* f21 = (const float4*)(base + ((size_t)x0 * GW + y1) * CD);
        const float4* f22 = (const float4*)(base + ((size_t)x1 * GW + y1) * CD);
        float w11 = dxw * dyw, w12 = (1.0f - dxw) * dyw;
        float w21 = dxw * (1.0f - dyw), w22 = (1.0f - dxw) * (1.0f - dyw);
#pragma unroll
        for (int t = 0; t < 4; ++t) {
            float4 a = f11[t], bb = f12[t], cc = f21[t], dd = f22[t];
            acc[4 * t + 0] += a.x * w11 + bb.x * w12 + cc.x * w21 + dd.x * w22;
            acc[4 * t + 1] += a.y * w11 + bb.y * w12 + cc.y * w21 + dd.y * w22;
            acc[4 * t + 2] += a.z * w11 + bb.z * w12 + cc.z * w21 + dd.z * w22;
            acc[4 * t + 3] += a.w * w11 + bb.w * w12 + cc.w * w21 + dd.w * w22;
        }
    }
#pragma unroll
    for (int k = 0; k < 16; ++k) fS[mIdx * 32 + hsel * 16 + k] = acc[k];
    DS_WAIT();
    // feat in C-fragment layout (lane = N, VGPR = M) for residual adds
    v8f featf0, featf1;
#pragma unroll
    for (int v = 0; v < 8; ++v) {
        int row = v + 8 * hsel;
        featf0[v] = fS[row * 32 + mIdx];
        featf1[v] = fS[row * 32 + 16 + mIdx];
    }

    // ---- layer 1: net = p @ fc_p_w.T + b  (K=3 zero-padded to 32) ----
    U16bf apU;
#pragma unroll
    for (int j = 0; j < 8; ++j) apU.u[j] = 0;
    if (hsel == 0) {            // lanes 0-15 hold K=0..2 (rest zero)
        apU.u[0] = packbf(px, py);
        apU.u[1] = packbf(pz, 0.0f);
    }
#pragma unroll
    for (int nt = 0; nt < 16; ++nt) {
        U16bf bU;
#pragma unroll
        for (int j = 0; j < 8; ++j) bU.u[j] = 0;
        if (hsel == 0) {
            const float* Wp = fc_p_w + (nt * 16 + mIdx) * 3;
            bU.u[0] = packbf(Wp[0], Wp[1]);
            bU.u[1] = packbf(Wp[2], 0.0f);
        }
        v8f cf = {};
        cf = wmma_bf16(apU.v, bU.v, cf);
        float bias = fc_p_b[nt * 16 + mIdx];
#pragma unroll
        for (int v = 0; v < 8; ++v)
            nS[(v + 8 * hsel) * 256 + nt * 16 + mIdx] = bfb(cf[v] + bias);
    }
    DS_WAIT();

    // ---- block0: h = relu(net)@w0.T + b0 ; sc = net@ws.T  (K=256, fused pass) ----
    v8f h0 = {}, h1 = {}, s0 = {}, s1 = {};
#pragma unroll
    for (int kt = 0; kt < 8; ++kt) {
        unsigned int r[8];
        loadRaw8(nS, mIdx, 128, kt, hsel, r);
        v16bf aN = mkA(r), aR = mkA_relu(r);
        v16bf b0f = loadBFragG(wfrag,        kt * 2 + 0, lane);
        v16bf b1f = loadBFragG(wfrag,        kt * 2 + 1, lane);
        v16bf bs0 = loadBFragG(wfrag + 4096, kt * 2 + 0, lane);
        v16bf bs1 = loadBFragG(wfrag + 4096, kt * 2 + 1, lane);
        h0 = wmma_bf16(aR, b0f, h0);
        h1 = wmma_bf16(aR, b1f, h1);
        s0 = wmma_bf16(aN, bs0, s0);
        s1 = wmma_bf16(aN, bs1, s1);
    }
    {
        float bb0 = blk0_b0[mIdx], bb1 = blk0_b0[16 + mIdx];
#pragma unroll
        for (int v = 0; v < 8; ++v) { h0[v] += bb0; h1[v] += bb1; }
    }
    storeC_relu_bf16(xw, h0, h1, mIdx, hsel);   // stage relu(h)
    DS_WAIT();
    v8f n0, n1;
    {
        unsigned int r[8];
        loadRaw8(xw, mIdx, 16, 0, hsel, r);
        v16bf aH = mkA(r);                      // already relu'd
        v8f d0 = {}, d1 = {};
        d0 = wmma_bf16(aH, loadBFragG(wfrag + 8192, 0, lane), d0);
        d1 = wmma_bf16(aH, loadBFragG(wfrag + 8192, 1, lane), d1);
        float bb0 = blk0_b1[mIdx], bb1 = blk0_b1[16 + mIdx];
#pragma unroll
        for (int v = 0; v < 8; ++v) {
            n0[v] = s0[v] + d0[v] + bb0 + featf0[v];
            n1[v] = s1[v] + d1[v] + bb1 + featf1[v];
        }
    }

    // ---- blocks 1..4 (32 -> 32, identity shortcut, + feat) ----
#pragma unroll
    for (int blk = 0; blk < 4; ++blk) {
        DS_WAIT();
        storeC_relu_bf16(xw, n0, n1, mIdx, hsel);   // stage relu(net)
        DS_WAIT();
        unsigned int r[8];
        loadRaw8(xw, mIdx, 16, 0, hsel, r);
        v16bf aR = mkA(r);
        v8f hh0 = {}, hh1 = {};
        hh0 = wmma_bf16(aR, loadBFragG(wfrag + 8704, blk * 2 + 0, lane), hh0);
        hh1 = wmma_bf16(aR, loadBFragG(wfrag + 8704, blk * 2 + 1, lane), hh1);
        float bb0 = blk_b0[blk * 32 + mIdx], bb1 = blk_b0[blk * 32 + 16 + mIdx];
#pragma unroll
        for (int v = 0; v < 8; ++v) { hh0[v] += bb0; hh1[v] += bb1; }
        storeC_relu_bf16(xw, hh0, hh1, mIdx, hsel); // stage relu(h)
        DS_WAIT();
        unsigned int r2[8];
        loadRaw8(xw, mIdx, 16, 0, hsel, r2);
        v16bf aH = mkA(r2);
        v8f d0 = {}, d1 = {};
        d0 = wmma_bf16(aH, loadBFragG(wfrag + 10752, blk * 2 + 0, lane), d0);
        d1 = wmma_bf16(aH, loadBFragG(wfrag + 10752, blk * 2 + 1, lane), d1);
        float cb0 = blk_b1[blk * 32 + mIdx], cb1 = blk_b1[blk * 32 + 16 + mIdx];
#pragma unroll
        for (int v = 0; v < 8; ++v) {
            n0[v] += d0[v] + cb0 + featf0[v];
            n1[v] += d1[v] + cb1 + featf1[v];
        }
    }

    // ---- out = relu(net) @ fc_out_w.T + b  (32 -> 1, shuffle reduce) ----
    float w0o = fc_out_w[mIdx], w1o = fc_out_w[16 + mIdx];
    float ob = fc_out_b[0];
#pragma unroll
    for (int v = 0; v < 8; ++v) {
        float pv = fmaxf(n0[v], 0.0f) * w0o + fmaxf(n1[v], 0.0f) * w1o;
#pragma unroll
        for (int off = 8; off >= 1; off >>= 1) pv += __shfl_xor(pv, off, 16);
        int o = ptBase + v + 8 * hsel;
        if (mIdx == 0 && o < nPts) out[o] = pv + ob;
    }
}

extern "C" void kernel_launch(void* const* d_in, const int* in_sizes, int n_in,
                              void* d_out, int out_size, void* d_ws, size_t ws_size,
                              hipStream_t stream) {
    const float* p        = (const float*)d_in[0];
    // d_in[1] = z (unused by reference)
    const float* c        = (const float*)d_in[2];
    const float* Cmat     = (const float*)d_in[3];
    const float* fc_p_w   = (const float*)d_in[4];
    const float* fc_p_b   = (const float*)d_in[5];
    const float* blk0_w0  = (const float*)d_in[6];
    const float* blk0_b0  = (const float*)d_in[7];
    const float* blk0_w1  = (const float*)d_in[8];
    const float* blk0_b1  = (const float*)d_in[9];
    const float* blk0_ws  = (const float*)d_in[10];
    const float* blk_w0   = (const float*)d_in[11];
    const float* blk_b0   = (const float*)d_in[12];
    const float* blk_w1   = (const float*)d_in[13];
    const float* blk_b1   = (const float*)d_in[14];
    const float* fc_out_w = (const float*)d_in[15];
    const float* fc_out_b = (const float*)d_in[16];
    unsigned int* wfrag = (unsigned int*)d_ws;   // 51200 B of bf16 B-fragments

    prep_kernel<<<1, 256, 0, stream>>>(blk0_w0, blk0_ws, blk0_w1, blk_w0, blk_w1, wfrag);

    int grid = (out_size + WAVES * 16 - 1) / (WAVES * 16);   // 3750 for 240000 pts
    decoder_kernel<<<grid, WAVES * 32, 0, stream>>>(
        p, c, Cmat, fc_p_w, fc_p_b, blk0_b0, blk0_b1, blk_b0, blk_b1,
        fc_out_w, fc_out_b, wfrag, (float*)d_out, out_size);
}
